// Multi_Head_Q_Shuffle_52029233823806
// MI455X (gfx1250) — matrix operand order repaired and checked by reading, not compiled, not run on hardware
//
#include <hip/hip_runtime.h>
#include <hip/hip_bf16.h>

#define BB 16
#define LL 1024
#define DD 512
#define HH 8
#define EE 64

typedef _Float16 f16;
typedef __attribute__((ext_vector_type(8)))  _Float16 v8h;
typedef __attribute__((ext_vector_type(16))) _Float16 v16h;
typedef __attribute__((ext_vector_type(8)))  float    v8f;

__device__ __forceinline__ v8f wmma_f16(v16h a, v16h b, v8f c) {
  return __builtin_amdgcn_wmma_f32_16x16x32_f16(
      false, a, false, b, (short)0, c, false, false);
}
__device__ __forceinline__ v16h cat8(v8h a, v8h b) {
  return __builtin_shufflevector(a, b, 0,1,2,3,4,5,6,7,8,9,10,11,12,13,14,15);
}
// A-fragment (16x32): row ptr pre-offset to row m = lane&15; hi = lane>>4.
// lanes<16: k = 0..15 ; lanes>=16: k = 16..31  (contiguous layout)
__device__ __forceinline__ v16h load_a16(const f16* row, int hi) {
  v8h lo = *(const v8h*)(row + hi * 16);
  v8h hh = *(const v8h*)(row + hi * 16 + 8);
  return cat8(lo, hh);
}
__device__ __forceinline__ v16h load16(const f16* p) { return *(const v16h*)p; }

// ---------------------------------------------------------------------------
__global__ void cast_f32_to_f16(const float* __restrict__ src,
                                f16* __restrict__ dst, int n) {
  int i = blockIdx.x * blockDim.x + threadIdx.x;
  if (i < n) dst[i] = (f16)src[i];
}

// Pack Wq/Wk/Wv [H,D,E] into B-fragment layout:
// idx = i + 16*(lane + 32*(nt + 4*(kt + 16*(h + 8*z))))
__global__ void pack_w(const float* __restrict__ Wq, const float* __restrict__ Wk,
                       const float* __restrict__ Wv, f16* __restrict__ whf) {
  int idx = blockIdx.x * 256 + threadIdx.x;
  if (idx >= 3 * HH * DD * EE) return;
  int i    = idx & 15;
  int lane = (idx >> 4) & 31;
  int nt   = (idx >> 9) & 3;
  int kt   = (idx >> 11) & 15;
  int h    = (idx >> 15) & 7;
  int z    = idx >> 18;
  int k = kt * 32 + (lane >> 4) * 16 + i;
  int n = nt * 16 + (lane & 15);
  const float* W = (z == 0) ? Wq : (z == 1) ? Wk : Wv;
  whf[idx] = (f16)W[((size_t)h * DD + k) * EE + n];
}

// Pack Wo [DIM, H*E]: B(k=i, n=d) = Wo[d][i]
__global__ void pack_wo(const float* __restrict__ Wo, f16* __restrict__ woh) {
  int idx = blockIdx.x * 256 + threadIdx.x;
  if (idx >= DD * HH * EE) return;
  int i    = idx & 15;
  int lane = (idx >> 4) & 31;
  int nt   = (idx >> 9) & 3;
  int kt   = (idx >> 11) & 15;
  int nb   = idx >> 15;
  int d = nb * 64 + nt * 16 + (lane & 15);
  int k = kt * 32 + (lane >> 4) * 16 + i;
  woh[idx] = (f16)Wo[(size_t)d * DD + k];
}

// ---------------------------------------------------------------------------
// QKV projection, software-pipelined + fully unrolled (copies become renames).
// z=0:Q (gathered A for h>0), z=1:K (row-major), z=2:V (transposed [E][L]).
// ---------------------------------------------------------------------------
__global__ __launch_bounds__(32)
void qkv_gemm(const f16* __restrict__ xh, const float* __restrict__ x,
              const int* __restrict__ perms, const f16* __restrict__ whf,
              f16* __restrict__ qkv) {
  const int z    = blockIdx.z;
  const int bh   = blockIdx.y;
  const int b    = bh >> 3, h = bh & 7;
  const int l0   = blockIdx.x * 16;
  const int lane = threadIdx.x & 31;
  const int n    = lane & 15, hi = lane >> 4, mb = hi * 8;
  const size_t HLE = (size_t)BB * HH * LL * EE;

  v8f acc[4] = {};
  const f16*  wfrag   = whf + ((size_t)(z * HH + h) * 16) * 4 * 32 * 16;
  const bool  gatherA = (z == 0) && (h != 0);
  const float* xb     = x + (size_t)b * LL * DD;
  const int*  perm    = perms + (size_t)(h > 0 ? h - 1 : 0) * LL * DD;
  const f16*  xa      = xh + ((size_t)b * LL + l0 + n) * DD;

  auto loadA = [&](int kt) -> v16h {
    if (gatherA) {
      const int rowbase = (l0 + n) * DD + kt * 32;
      v16h a;
#pragma unroll
      for (int i = 0; i < 8; ++i) {
        a[i]     = (f16)xb[perm[rowbase + hi * 16 + i]];
        a[8 + i] = (f16)xb[perm[rowbase + hi * 16 + 8 + i]];
      }
      return a;
    }
    return load_a16(xa + kt * 32, hi);
  };

  v16h a_cur = loadA(0);
  v16h b_cur[4];
#pragma unroll
  for (int nt = 0; nt < 4; ++nt)
    b_cur[nt] = load16(wfrag + ((size_t)nt * 32 + lane) * 16);

#pragma unroll
  for (int kt = 0; kt < 15; ++kt) {
    v16h a_nxt = loadA(kt + 1);
    v16h b_nxt[4];
#pragma unroll
    for (int nt = 0; nt < 4; ++nt)
      b_nxt[nt] = load16(wfrag + ((size_t)((kt + 1) * 4 + nt) * 32 + lane) * 16);
#pragma unroll
    for (int nt = 0; nt < 4; ++nt) acc[nt] = wmma_f16(a_cur, b_cur[nt], acc[nt]);
    a_cur = a_nxt;
#pragma unroll
    for (int nt = 0; nt < 4; ++nt) b_cur[nt] = b_nxt[nt];
  }
#pragma unroll
  for (int nt = 0; nt < 4; ++nt) acc[nt] = wmma_f16(a_cur, b_cur[nt], acc[nt]);

  if (z == 2) {
    f16* vtp = qkv + 2 * HLE + (size_t)bh * EE * LL;   // V^T [E][L]
#pragma unroll
    for (int nt = 0; nt < 4; ++nt) {
      v8h pk;
#pragma unroll
      for (int r = 0; r < 8; ++r) pk[r] = (f16)acc[nt][r];
      *(v8h*)(vtp + (size_t)(nt * 16 + n) * LL + l0 + mb) = pk;
    }
  } else {
    f16* out = qkv + (size_t)z * HLE + ((size_t)bh * LL + l0) * EE;
#pragma unroll
    for (int nt = 0; nt < 4; ++nt)
#pragma unroll
      for (int r = 0; r < 8; ++r)
        out[(size_t)(mb + r) * EE + nt * 16 + n] = (f16)acc[nt][r];
  }
}

// ---------------------------------------------------------------------------
// Flash attention: one wave = 32 query rows; each lane owns one softmax row.
// V fragments issued early so score-WMMAs + softmax hide their latency.
// ---------------------------------------------------------------------------
__global__ __launch_bounds__(32)
void flash_attn(const f16* __restrict__ qkv, f16* __restrict__ ctx) {
  __shared__ float s_s[32][33];
  __shared__ f16   s_p[32][32];
  __shared__ float s_alpha[32], s_rsum[32];

  const int bh   = blockIdx.y, b = bh >> 3, h = bh & 7;
  const int l0   = blockIdx.x * 32;
  const int lane = threadIdx.x & 31;
  const int n    = lane & 15, hi = lane >> 4, mb = hi * 8;

  const size_t HLE = (size_t)BB * HH * LL * EE;
  const f16* qp = qkv + ((size_t)bh * LL + l0) * EE;
  const f16* kp = qkv + HLE + (size_t)bh * LL * EE;
  const f16* vt = qkv + 2 * HLE + (size_t)bh * EE * LL;   // [E][L]

  v16h aq[2][2];
#pragma unroll
  for (int mt = 0; mt < 2; ++mt)
#pragma unroll
    for (int j = 0; j < 2; ++j)
      aq[mt][j] = load_a16(qp + (size_t)(mt * 16 + n) * EE + j * 32, hi);

  float rmax = -3.0e38f, rsum = 0.0f;
  v8f acc[2][4] = {};
  const float scale = 0.125f;   // 64^-0.5

  for (int kt = 0; kt < LL; kt += 32) {
    if (kt + 32 < LL) {
      __builtin_prefetch(kp + (size_t)(kt + 32 + n) * EE, 0, 1);
      __builtin_prefetch(vt + (size_t)(n * 4) * LL + kt + 32, 0, 1);
    }
    // ---- V fragments for this chunk, issued early ----
    v16h bv[4];
#pragma unroll
    for (int nt = 0; nt < 4; ++nt)
      bv[nt] = load16(vt + (size_t)(nt * 16 + n) * LL + kt + hi * 16);
    // ---- K fragments (one clause), then 8 score WMMAs ----
    v16h bk[4];   // [j*2 + keyhalf]
#pragma unroll
    for (int j = 0; j < 2; ++j) {
      bk[j * 2 + 0] = load16(kp + (size_t)(kt + n) * EE + j * 32 + hi * 16);
      bk[j * 2 + 1] = load16(kp + (size_t)(kt + 16 + n) * EE + j * 32 + hi * 16);
    }
    v8f s[2][2] = {};
#pragma unroll
    for (int j = 0; j < 2; ++j)
#pragma unroll
      for (int mt = 0; mt < 2; ++mt) {
        s[mt][0] = wmma_f16(aq[mt][j], bk[j * 2 + 0], s[mt][0]);
        s[mt][1] = wmma_f16(aq[mt][j], bk[j * 2 + 1], s[mt][1]);
      }
#pragma unroll
    for (int mt = 0; mt < 2; ++mt)
#pragma unroll
      for (int r = 0; r < 8; ++r) {
        s_s[mt * 16 + mb + r][n]      = s[mt][0][r] * scale;
        s_s[mt * 16 + mb + r][16 + n] = s[mt][1][r] * scale;
      }
    __syncthreads();
    // ---- online softmax: lane owns row `lane` ----
    {
      float cmax = rmax;
#pragma unroll
      for (int c = 0; c < 32; ++c) cmax = fmaxf(cmax, s_s[lane][c]);
      float alpha = __expf(rmax - cmax);
      float csum = 0.0f;
#pragma unroll
      for (int c = 0; c < 32; ++c) {
        float p = __expf(s_s[lane][c] - cmax);
        s_p[lane][c] = (f16)p;
        csum += p;
      }
      rmax = cmax;
      rsum = rsum * alpha + csum;
      s_alpha[lane] = alpha;
    }
    __syncthreads();
    // ---- rescale running O ----
#pragma unroll
    for (int mt = 0; mt < 2; ++mt)
#pragma unroll
      for (int r = 0; r < 8; ++r) {
        float al = s_alpha[mt * 16 + mb + r];
#pragma unroll
        for (int nt = 0; nt < 4; ++nt) acc[mt][nt][r] *= al;
      }
    // ---- P fragments from LDS; O += P @ V ----
    v16h ap[2];
#pragma unroll
    for (int mt = 0; mt < 2; ++mt) ap[mt] = load_a16(&s_p[mt * 16 + n][0], hi);
#pragma unroll
    for (int nt = 0; nt < 4; ++nt)
#pragma unroll
      for (int mt = 0; mt < 2; ++mt)
        acc[mt][nt] = wmma_f16(ap[mt], bv[nt], acc[mt][nt]);
    __syncthreads();
  }
  // ---- normalize, store head-concatenated context [B, L, H*E] ----
  s_rsum[lane] = rsum;
  __syncthreads();
  f16* op = ctx + ((size_t)b * LL + l0) * (HH * EE) + h * EE;
#pragma unroll
  for (int mt = 0; mt < 2; ++mt)
#pragma unroll
    for (int r = 0; r < 8; ++r) {
      float inv = 1.0f / s_rsum[mt * 16 + mb + r];
#pragma unroll
      for (int nt = 0; nt < 4; ++nt)
        op[(size_t)(mt * 16 + mb + r) * (HH * EE) + nt * 16 + n] =
            (f16)(acc[mt][nt][r] * inv);
    }
}

// ---------------------------------------------------------------------------
// Output projection, software-pipelined + fully unrolled.
// ---------------------------------------------------------------------------
__global__ __launch_bounds__(32)
void out_gemm(const f16* __restrict__ ctx, const f16* __restrict__ woh,
              const float* __restrict__ bo, float* __restrict__ out) {
  const int m0   = blockIdx.x * 16;
  const int nb   = blockIdx.y;
  const int lane = threadIdx.x & 31;
  const int n    = lane & 15, hi = lane >> 4, mb = hi * 8;

  v8f acc[4] = {};
  const f16* wfrag = woh + (size_t)nb * 16 * 4 * 32 * 16;
  const f16* ar    = ctx + (size_t)(m0 + n) * DD;

  v16h a_cur = load_a16(ar, hi);
  v16h b_cur[4];
#pragma unroll
  for (int nt = 0; nt < 4; ++nt)
    b_cur[nt] = load16(wfrag + ((size_t)nt * 32 + lane) * 16);

#pragma unroll
  for (int kt = 0; kt < 15; ++kt) {
    v16h a_nxt = load_a16(ar + (kt + 1) * 32, hi);
    v16h b_nxt[4];
#pragma unroll
    for (int nt = 0; nt < 4; ++nt)
      b_nxt[nt] = load16(wfrag + ((size_t)((kt + 1) * 4 + nt) * 32 + lane) * 16);
#pragma unroll
    for (int nt = 0; nt < 4; ++nt) acc[nt] = wmma_f16(a_cur, b_cur[nt], acc[nt]);
    a_cur = a_nxt;
#pragma unroll
    for (int nt = 0; nt < 4; ++nt) b_cur[nt] = b_nxt[nt];
  }
#pragma unroll
  for (int nt = 0; nt < 4; ++nt) acc[nt] = wmma_f16(a_cur, b_cur[nt], acc[nt]);

#pragma unroll
  for (int nt = 0; nt < 4; ++nt) {
    float bias = bo[nb * 64 + nt * 16 + n];
#pragma unroll
    for (int r = 0; r < 8; ++r)
      out[(size_t)(m0 + mb + r) * DD + nb * 64 + nt * 16 + n] = acc[nt][r] + bias;
  }
}

// ---------------------------------------------------------------------------
extern "C" void kernel_launch(void* const* d_in, const int* in_sizes, int n_in,
                              void* d_out, int out_size, void* d_ws, size_t ws_size,
                              hipStream_t stream) {
  const float* x  = (const float*)d_in[0];
  const int*   pm = (const int*)d_in[1];
  const float* Wq = (const float*)d_in[2];
  const float* Wk = (const float*)d_in[3];
  const float* Wv = (const float*)d_in[4];
  const float* Wo = (const float*)d_in[5];
  const float* bo = (const float*)d_in[6];
  float* out = (float*)d_out;

  f16* xh  = (f16*)d_ws;
  f16* whf = xh  + (size_t)BB * LL * DD;
  f16* woh = whf + (size_t)3 * HH * DD * EE;
  f16* qkv = woh + (size_t)DD * HH * EE;
  f16* ctx = qkv + (size_t)3 * BB * HH * LL * EE;

  const int nx  = BB * LL * DD;
  const int nw3 = 3 * HH * DD * EE;
  const int nwo = DD * HH * EE;
  cast_f32_to_f16<<<(nx + 255) / 256, 256, 0, stream>>>(x, xh, nx);
  pack_w <<<(nw3 + 255) / 256, 256, 0, stream>>>(Wq, Wk, Wv, whf);
  pack_wo<<<(nwo + 255) / 256, 256, 0, stream>>>(Wo, woh);

  qkv_gemm  <<<dim3(LL / 16, BB * HH, 3), 32, 0, stream>>>(xh, x, pm, whf, qkv);
  flash_attn<<<dim3(LL / 32, BB * HH),    32, 0, stream>>>(qkv, ctx);
  out_gemm  <<<dim3(BB * LL / 16, DD / 64), 32, 0, stream>>>(ctx, woh, bo, out);
}
